// RelationalTransformerUpdate_17016660426889
// MI455X (gfx1250) — compile-verified, matrix operand order checked by reading, not run on hardware
//
#include <hip/hip_runtime.h>

// Problem constants (fixed by the reference)
#define NTOK 1024
#define HID  512
#define NHEAD 8
#define DKD  64
#define FFD  2048
#define NREL 37
#define NLAYER 4
#define KCHUNK 64

typedef __bf16 bhalf;
typedef __attribute__((ext_vector_type(16))) __bf16 v16bf;
typedef __attribute__((ext_vector_type(8)))  __bf16 v8bf;
typedef __attribute__((ext_vector_type(8)))  float  v8f;
typedef __attribute__((ext_vector_type(4)))  int    v4i;

// ---- gfx1250 async global->LDS copy (guarded; falls back to sync staging) ----
#if defined(__HIP_DEVICE_COMPILE__) && \
    __has_builtin(__builtin_amdgcn_global_load_async_to_lds_b128)
#define ASYNC_LDS 1
#endif

__device__ __forceinline__ void cp16_to_lds(const bhalf* g, bhalf* l) {
#ifdef ASYNC_LDS
  __builtin_amdgcn_global_load_async_to_lds_b128(
      (__attribute__((address_space(1))) v4i*)g,
      (__attribute__((address_space(3))) v4i*)l, 0, 0);
#else
  *(uint4*)l = *(const uint4*)g;
#endif
}

__device__ __forceinline__ void wait_async_copies() {
#ifdef ASYNC_LDS
#if __has_builtin(__builtin_amdgcn_s_wait_asynccnt)
  __builtin_amdgcn_s_wait_asynccnt(0);
#else
  asm volatile("s_wait_asynccnt 0" ::: "memory");
#endif
#endif
}

// Assemble one 16-lane-row WMMA operand fragment: two contiguous 16B runs.
// CDNA5 16-bit A/B layout: lane L holds row (L&15); lanes 0-15 carry
// K = {0..7, 16..23}, lanes 16-31 carry K = {8..15, 24..31}.
__device__ __forceinline__ v16bf load_frag16(const bhalf* p) {
  v8bf lo = *(const v8bf*)(p);
  v8bf hi = *(const v8bf*)(p + 16);
  return __builtin_shufflevector(lo, hi, 0,1,2,3,4,5,6,7,8,9,10,11,12,13,14,15);
}

struct GemmCfg {
  const bhalf* A; long lda;     // A: [M,K] bf16 row-major
  const bhalf* B; long ldb;     // B: [Ncols,K] bf16 (pre-transposed: row = output col)
  const float* bias;            // [Ncols] or null
  const float* res;             // residual, ldc layout, or null
  float* outF;                  // f32 output, or null
  bhalf* outB;                  // bf16 output, or null
  bhalf* outBT; long ldbt;      // bf16 transposed output outBT[col*ldbt+row], or null
  long ldc;
  int  K;                       // multiple of KCHUNK
  int  relu;
};

// bf16 WMMA GEMM, LDS double-buffered: 256 threads = 8 wave32 waves, wave grid
// 2(M)x4(N) of 16x16 tiles -> block tile 32x64. K staged in 64-element chunks
// via async global->LDS copies; fragments consumed with ds_load_b128.
// grid = (Ncols/64, M/32). LDS: 2*(32*64 + 64*64)*2B = 24 KB.
__global__ __launch_bounds__(256) void wmma_gemm_kernel(GemmCfg c) {
  __shared__ __align__(16) bhalf sA[2][32 * KCHUNK];
  __shared__ __align__(16) bhalf sB[2][64 * KCHUNK];

  const int t    = threadIdx.x;
  const int lane = t & 31;
  const int wave = t >> 5;
  const long rowBlk = (long)blockIdx.y * 32;
  const long colBlk = (long)blockIdx.x * 64;

  // tile-copy assignment: 16B chunk per thread for A, two for B
  const int cr  = t >> 3;          // 0..31
  const int ck  = (t & 7) << 3;    // 0,8,...,56
  const bhalf* gA  = c.A + (rowBlk + cr) * c.lda + ck;
  const bhalf* gB0 = c.B + (colBlk + cr) * c.ldb + ck;
  const bhalf* gB1 = c.B + (colBlk + 32 + cr) * c.ldb + ck;
  const int lA  = cr * KCHUNK + ck;
  const int lB0 = cr * KCHUNK + ck;
  const int lB1 = (32 + cr) * KCHUNK + ck;

  // per-lane fragment offsets inside the LDS tiles
  const int idx15 = lane & 15;
  const int kb    = (lane >> 4) << 3;     // 0 or 8
  const int aFrag = ((wave >> 2) * 16 + idx15) * KCHUNK + kb;
  const int bFrag = ((wave & 3)  * 16 + idx15) * KCHUNK + kb;

  const int nCh = c.K / KCHUNK;
  // prologue: stage chunk 0 into buffer 0
  cp16_to_lds(gA, &sA[0][lA]);
  cp16_to_lds(gB0, &sB[0][lB0]);
  cp16_to_lds(gB1, &sB[0][lB1]);

  v8f acc = {};
  for (int kc = 0; kc < nCh; ++kc) {
    const int cur = kc & 1;
    wait_async_copies();
    __syncthreads();
    if (kc + 1 < nCh) {                 // overlap next chunk's copy with math
      const int k0 = (kc + 1) * KCHUNK;
      cp16_to_lds(gA + k0, &sA[1 - cur][lA]);
      cp16_to_lds(gB0 + k0, &sB[1 - cur][lB0]);
      cp16_to_lds(gB1 + k0, &sB[1 - cur][lB1]);
    }
    const bhalf* la = &sA[cur][aFrag];
    const bhalf* lb = &sB[cur][bFrag];
#pragma unroll
    for (int ks = 0; ks < KCHUNK; ks += 32) {
      v16bf fa = load_frag16(la + ks);
      v16bf fb = load_frag16(lb + ks);
      acc = __builtin_amdgcn_wmma_f32_16x16x32_bf16(
          /*neg_a=*/false, fa, /*neg_b=*/false, fb,
          /*c_mod=*/(short)0, acc, /*reuse_a=*/false, /*reuse_b=*/false);
    }
    __syncthreads();                    // reads done before buffer is rewritten
  }

  // C/D layout: VGPR v holds (M = v + 8*(lane>>4), N = lane&15)
  const long rowBase = rowBlk + (long)(wave >> 2) * 16;
  const long col     = colBlk + (long)(wave & 3) * 16 + idx15;
  const int  mb      = (lane >> 4) << 3;
  const float bv = c.bias ? c.bias[col] : 0.0f;
#pragma unroll
  for (int v = 0; v < 8; ++v) {
    const long row = rowBase + mb + v;
    float val = acc[v] + bv;
    if (c.relu) val = fmaxf(val, 0.0f);
    const long ci = row * c.ldc + col;
    if (c.res)   val += c.res[ci];
    if (c.outF)  c.outF[ci] = val;
    if (c.outB)  c.outB[ci] = (bhalf)val;
    if (c.outBT) c.outBT[col * c.ldbt + row] = (bhalf)val;
  }
}

// LayerNorm over H=512, one block (256 threads) per row; out f32 or bf16.
template <typename OutT>
__global__ __launch_bounds__(256) void ln_kernel(const float* __restrict__ x,
                                                 const float* __restrict__ g,
                                                 const float* __restrict__ b,
                                                 OutT* __restrict__ y) {
  __shared__ float red[256];
  const int i = blockIdx.x, t = threadIdx.x;
  const float* xr = x + (size_t)i * HID;
  const float v0 = xr[t], v1 = xr[t + 256];
  red[t] = v0 + v1; __syncthreads();
  for (int o = 128; o > 0; o >>= 1) { if (t < o) red[t] += red[t + o]; __syncthreads(); }
  const float mean = red[0] * (1.0f / HID); __syncthreads();
  red[t] = v0 * v0 + v1 * v1; __syncthreads();
  for (int o = 128; o > 0; o >>= 1) { if (t < o) red[t] += red[t + o]; __syncthreads(); }
  const float var = red[0] * (1.0f / HID) - mean * mean;
  const float rstd = rsqrtf(var + 1e-6f);
  y[(size_t)i * HID + t]       = (OutT)(g[t] * (v0 - mean) * rstd + b[t]);
  y[(size_t)i * HID + t + 256] = (OutT)(g[t + 256] * (v1 - mean) * rstd + b[t + 256]);
}

// W [K,Ncol] f32 -> WT [Ncol,K] bf16 (transpose so GEMM B-frag loads are contiguous)
__global__ void convT_kernel(const float* __restrict__ W, bhalf* __restrict__ WT,
                             int K, int Ncol) {
  size_t t = (size_t)blockIdx.x * 256 + threadIdx.x;
  if (t >= (size_t)K * Ncol) return;
  const int k = (int)(t / Ncol), c = (int)(t % Ncol);
  WT[(size_t)c * K + k] = (bhalf)W[t];
}

// rel [NREL,64] f32 -> out [64,64] bf16, zero-padded; optionally transposed.
__global__ void rel_pad_kernel(const float* __restrict__ rel, bhalf* __restrict__ out,
                               int transpose) {
  const int t = blockIdx.x * 256 + threadIdx.x;   // 4096 total
  const int r = t >> 6, c = t & 63;
  float v;
  if (!transpose) v = (r < NREL) ? rel[r * 64 + c] : 0.0f;  // out[r][d] = rel[r][d]
  else            v = (c < NREL) ? rel[c * 64 + r] : 0.0f;  // out[d][r] = rel[r][d]
  out[t] = (bhalf)v;
}

// Per-head fused: scores + relation bias -> softmax -> bf16 probs + per-relation
// bucketed prob sums (for the a@rel_v term). One block per query row.
__global__ __launch_bounds__(256) void softmax_rel_kernel(
    const float* __restrict__ scores,   // [N,N]
    const float* __restrict__ qrel,     // [N,64] (cols >= NREL are zero)
    const int*   __restrict__ rids,     // [N,N]
    bhalf* __restrict__ aOut,           // [N,N] bf16
    bhalf* __restrict__ arelOut) {      // [N,64] bf16 (padded)
  __shared__ float red[256];
  __shared__ float srel[64];
  const int i = blockIdx.x, t = threadIdx.x;
  if (t < 64) srel[t] = 0.0f;
  const float* qr = qrel + (size_t)i * 64;
  const float scale = 0.125f;            // 1/sqrt(64)
  float s[4]; int rid[4];
  float lmax = -3.4e38f;
#pragma unroll
  for (int u = 0; u < 4; ++u) {
    const int j = t + u * 256;
    rid[u] = rids[(size_t)i * NTOK + j];
    s[u] = (scores[(size_t)i * NTOK + j] + qr[rid[u]]) * scale;
    lmax = fmaxf(lmax, s[u]);
  }
  red[t] = lmax; __syncthreads();
  for (int o = 128; o > 0; o >>= 1) { if (t < o) red[t] = fmaxf(red[t], red[t + o]); __syncthreads(); }
  const float m = red[0]; __syncthreads();
  float p[4], lsum = 0.0f;
#pragma unroll
  for (int u = 0; u < 4; ++u) { p[u] = __expf(s[u] - m); lsum += p[u]; }
  red[t] = lsum; __syncthreads();
  for (int o = 128; o > 0; o >>= 1) { if (t < o) red[t] += red[t + o]; __syncthreads(); }
  const float inv = 1.0f / red[0];
#pragma unroll
  for (int u = 0; u < 4; ++u) {
    const int j = t + u * 256;
    const float a = p[u] * inv;
    aOut[(size_t)i * NTOK + j] = (bhalf)a;
    atomicAdd(&srel[rid[u]], a);         // ds_add_f32
  }
  __syncthreads();
  if (t < 64) arelOut[(size_t)i * 64 + t] = (bhalf)srel[t];
}

static inline void launch_gemm(const GemmCfg& c, int M, int Ncols, hipStream_t s) {
  dim3 grid(Ncols / 64, M / 32);
  wmma_gemm_kernel<<<grid, dim3(256), 0, s>>>(c);
}

extern "C" void kernel_launch(void* const* d_in, const int* in_sizes, int n_in,
                              void* d_out, int out_size, void* d_ws, size_t ws_size,
                              hipStream_t stream) {
  (void)in_sizes; (void)n_in; (void)out_size; (void)ws_size;
  const float* x     = (const float*)d_in[0];
  const int*   rids  = (const int*)  d_in[1];
  const float* Wq = (const float*)d_in[2];  const float* bq = (const float*)d_in[3];
  const float* Wk = (const float*)d_in[4];  const float* bk = (const float*)d_in[5];
  const float* Wv = (const float*)d_in[6];  const float* bv = (const float*)d_in[7];
  const float* Wo = (const float*)d_in[8];  const float* bo = (const float*)d_in[9];
  const float* relK = (const float*)d_in[10];
  const float* relV = (const float*)d_in[11];
  const float* ln1g = (const float*)d_in[12]; const float* ln1b = (const float*)d_in[13];
  const float* ln2g = (const float*)d_in[14]; const float* ln2b = (const float*)d_in[15];
  const float* W1 = (const float*)d_in[16]; const float* b1 = (const float*)d_in[17];
  const float* W2 = (const float*)d_in[18]; const float* b2 = (const float*)d_in[19];
  const float* lnfg = (const float*)d_in[20]; const float* lnfb = (const float*)d_in[21];

  // ---- workspace carve (~27 MB) ----
  uintptr_t base = (uintptr_t)d_ws;
  auto carve = [&](size_t bytes) -> void* {
    void* r = (void*)base; base += (bytes + 255) & ~(size_t)255; return r;
  };
  float* xc     = (float*)carve((size_t)NTOK * HID * 4);
  bhalf* ybf    = (bhalf*)carve((size_t)NTOK * HID * 2);
  bhalf* qbf    = (bhalf*)carve((size_t)NTOK * HID * 2);
  bhalf* kbf    = (bhalf*)carve((size_t)NTOK * HID * 2);
  bhalf* vbf    = (bhalf*)carve((size_t)NTOK * HID * 2);
  bhalf* vTbf   = (bhalf*)carve((size_t)HID * NTOK * 2);
  bhalf* WqT    = (bhalf*)carve((size_t)HID * HID * 2);
  bhalf* WkT    = (bhalf*)carve((size_t)HID * HID * 2);
  bhalf* WvT    = (bhalf*)carve((size_t)HID * HID * 2);
  bhalf* WoT    = (bhalf*)carve((size_t)HID * HID * 2);
  bhalf* W1T    = (bhalf*)carve((size_t)FFD * HID * 2);
  bhalf* W2T    = (bhalf*)carve((size_t)HID * FFD * 2);
  bhalf* relKP  = (bhalf*)carve((size_t)64 * 64 * 2);
  bhalf* relVT  = (bhalf*)carve((size_t)64 * 64 * 2);
  float* qrelF  = (float*)carve((size_t)NTOK * 64 * 4);
  float* scores = (float*)carve((size_t)NTOK * NTOK * 4);
  bhalf* abf    = (bhalf*)carve((size_t)NTOK * NTOK * 2);
  bhalf* arelbf = (bhalf*)carve((size_t)NTOK * 64 * 2);
  float* oF     = (float*)carve((size_t)NTOK * HID * 4);
  bhalf* attnbf = (bhalf*)carve((size_t)NTOK * HID * 2);
  bhalf* h1bf   = (bhalf*)carve((size_t)NTOK * FFD * 2);

  // residual stream xc = x
  (void)hipMemcpyAsync(xc, x, (size_t)NTOK * HID * 4, hipMemcpyDeviceToDevice, stream);

  const int cvtHH = (HID * HID + 255) / 256;
  const int cvtHF = (HID * FFD + 255) / 256;

  for (int l = 0; l < NLAYER; ++l) {
    const size_t oHH = (size_t)l * HID * HID;
    const size_t oH  = (size_t)l * HID;
    const size_t oHF = (size_t)l * HID * FFD;
    const size_t oF_ = (size_t)l * FFD;
    const size_t oR  = (size_t)l * NREL * DKD;

    // weights -> transposed bf16
    convT_kernel<<<cvtHH, 256, 0, stream>>>(Wq + oHH, WqT, HID, HID);
    convT_kernel<<<cvtHH, 256, 0, stream>>>(Wk + oHH, WkT, HID, HID);
    convT_kernel<<<cvtHH, 256, 0, stream>>>(Wv + oHH, WvT, HID, HID);
    convT_kernel<<<cvtHH, 256, 0, stream>>>(Wo + oHH, WoT, HID, HID);
    convT_kernel<<<cvtHF, 256, 0, stream>>>(W1 + oHF, W1T, HID, FFD);
    convT_kernel<<<cvtHF, 256, 0, stream>>>(W2 + oHF, W2T, FFD, HID);
    rel_pad_kernel<<<16, 256, 0, stream>>>(relK + oR, relKP, 0);
    rel_pad_kernel<<<16, 256, 0, stream>>>(relV + oR, relVT, 1);

    // ---- attention sublayer ----
    ln_kernel<bhalf><<<NTOK, 256, 0, stream>>>(xc, ln1g + oH, ln1b + oH, ybf);

    GemmCfg g = {};
    // Q,K,V projections
    g = {ybf, HID, WqT, HID, bq + oH, nullptr, nullptr, qbf, nullptr, 0, HID, HID, 0};
    launch_gemm(g, NTOK, HID, stream);
    g = {ybf, HID, WkT, HID, bk + oH, nullptr, nullptr, kbf, nullptr, 0, HID, HID, 0};
    launch_gemm(g, NTOK, HID, stream);
    g = {ybf, HID, WvT, HID, bv + oH, nullptr, nullptr, vbf, vTbf, NTOK, HID, HID, 0};
    launch_gemm(g, NTOK, HID, stream);

    for (int h = 0; h < NHEAD; ++h) {
      const bhalf* qh = qbf + h * DKD;
      const bhalf* kh = kbf + h * DKD;
      // qrel[i,r] = q_h[i,:] . rel_k[r,:]   (R padded to 64)
      g = {qh, HID, relKP, 64, nullptr, nullptr, qrelF, nullptr, nullptr, 0, 64, DKD, 0};
      launch_gemm(g, NTOK, 64, stream);
      // scores = q_h @ k_h^T
      g = {qh, HID, kh, HID, nullptr, nullptr, scores, nullptr, nullptr, 0, NTOK, DKD, 0};
      launch_gemm(g, NTOK, NTOK, stream);
      // softmax with relation bias; emits bf16 probs + per-relation bucket sums
      softmax_rel_kernel<<<NTOK, 256, 0, stream>>>(scores, qrelF, rids, abf, arelbf);
      // o_h = A @ V_h  (B = V^T rows are contiguous)
      g = {abf, NTOK, vTbf + (size_t)h * DKD * NTOK, NTOK, nullptr, nullptr,
           oF + h * DKD, nullptr, nullptr, 0, HID, NTOK, 0};
      launch_gemm(g, NTOK, 64, stream);
      // o_h += arel @ rel_v ; also emit bf16 copy for the output projection
      g = {arelbf, 64, relVT, 64, nullptr, oF + h * DKD, oF + h * DKD,
           attnbf + h * DKD, nullptr, 0, HID, 64, 0};
      launch_gemm(g, NTOK, 64, stream);
    }
    // xc = xc + attn @ Wo + bo
    g = {attnbf, HID, WoT, HID, bo + oH, xc, xc, nullptr, nullptr, 0, HID, HID, 0};
    launch_gemm(g, NTOK, HID, stream);

    // ---- FFN sublayer ----
    ln_kernel<bhalf><<<NTOK, 256, 0, stream>>>(xc, ln2g + oH, ln2b + oH, ybf);
    g = {ybf, HID, W1T, HID, b1 + oF_, nullptr, nullptr, h1bf, nullptr, 0, FFD, HID, 1};
    launch_gemm(g, NTOK, FFD, stream);
    g = {h1bf, FFD, W2T, FFD, b2 + oH, xc, xc, nullptr, nullptr, 0, HID, FFD, 0};
    launch_gemm(g, NTOK, HID, stream);
  }

  // final LayerNorm -> d_out (f32)
  ln_kernel<float><<<NTOK, 256, 0, stream>>>(xc, lnfg, lnfb, (float*)d_out);
}
